// Decoder_19791209300188
// MI455X (gfx1250) — matrix operand order repaired and checked by reading, not compiled, not run on hardware
//
#include <hip/hip_runtime.h>
#include <hip/hip_bf16.h>
#include <stdint.h>

// ---------------------------------------------------------------------------
// Tacotron2 decoder for MI455X (gfx1250, wave32, WMMA bf16 16x16x32).
// GEMMs: 64x128 block tile, 8 waves, each wave a 16x64 strip (4 WMMAs/k-step),
// uint4-vectorized LDS staging with register double-buffering.
// ---------------------------------------------------------------------------

#define B_   64
#define T_   512
#define S_   500
#define M_   80
#define E_   512
#define P_   256
#define H_   1024
#define A_   128
#define FLOC 32
#define KLOC 31
#define NEG_INF (-1.0e9f)

typedef unsigned short u16;
typedef unsigned int   u32;

typedef __attribute__((ext_vector_type(16))) __bf16 v16bf;
typedef __attribute__((ext_vector_type(8)))  float  v8f;

union FragBF { v16bf v; u32 u[8]; };

#define DIVUP(a,b) (((a)+(b)-1)/(b))

// ---------------- helpers --------------------------------------------------
__device__ __forceinline__ u16 f2bf(float f) {           // round-to-nearest-even
  u32 u = __float_as_uint(f);
  u32 r = (u + 0x7FFFu + ((u >> 16) & 1u)) >> 16;
  return (u16)r;
}
__device__ __forceinline__ u32 pack2bf(float a, float b) {
  return (u32)f2bf(a) | ((u32)f2bf(b) << 16);
}
__device__ __forceinline__ u32 hashu(u32 x) {
  x ^= x >> 17; x *= 0xed5ad4bbu; x ^= x >> 11;
  x *= 0xac4c1b51u; x ^= x >> 15; x *= 0x31848babu; x ^= x >> 14;
  return x;
}
__device__ __forceinline__ float drop_scale(u32 seed, u32 idx, float p) {
  float r = (float)(hashu(seed ^ (idx * 0x9E3779B9u)) >> 8) * (1.0f / 16777216.0f);
  float keep = 1.0f - p;
  return (r < keep) ? (1.0f / keep) : 0.0f;
}
__device__ __forceinline__ float sigf(float x) { return 1.0f / (1.0f + __expf(-x)); }

// ---------------- WMMA GEMM:  C(rows x N) = A(rows x K) * W(N x K)^T --------
// A: bf16 (AF32=0) or fp32 converted in-flight (AF32=1); W: bf16 row-major.
// mode 0: fp32 out (+bias). mode 1: relu+dropout -> bf16 out.
// rows % 64 == 0, N % 128 == 0, K % 32 == 0.
// Block 256 threads (8 waves); tile 64x128; wave = 16 rows x 64 cols.
template <int AF32>
__global__ __launch_bounds__(256) void kGemmT(
    const void* __restrict__ Aptr,
    const u16* __restrict__ W, const float* __restrict__ bias,
    void* __restrict__ Cout, int K, int N, int mode, float dropP, u32 seed)
{
  __shared__ u32 lA[64 * 16];    // 64 rows x 32 bf16 (16 dwords)
  __shared__ u32 lB[128 * 16];   // 128 cols x 32 bf16

  const int row0 = blockIdx.x * 64;
  const int col0 = blockIdx.y * 128;
  const int tid  = threadIdx.x;
  const int lane = tid & 31;
  const int wid  = tid >> 5;
  const int mi   = wid & 3;          // row tile 0..3
  const int nj   = wid >> 2;         // col group 0..1 (64 cols each)
  const int lm   = lane & 15;
  const int h    = lane >> 4;
  const int Kd   = K >> 1;           // bf16-pair dwords per row

  const u32*   Au = (const u32*)Aptr;
  const float* Af = (const float*)Aptr;
  const u32*   Wu = (const u32*)W;

  // staging coordinates: 4 uint4 per 16-dword row
  const int arow = tid >> 2, ac4 = tid & 3;        // A: 64 rows
  const int brow0 = tid >> 2, brow1 = brow0 + 64;  // B: 128 rows, 2 per thread
  const int bc4 = tid & 3;

  uint4 ra, rb0, rb1;
  auto loadTiles = [&](int kk) {
    if (AF32) {
      const float4* p = (const float4*)(Af + (size_t)(row0 + arow) * K + kk);
      float4 x = p[ac4 * 2], y = p[ac4 * 2 + 1];
      ra = make_uint4(pack2bf(x.x, x.y), pack2bf(x.z, x.w),
                      pack2bf(y.x, y.y), pack2bf(y.z, y.w));
    } else {
      ra = ((const uint4*)(Au + (size_t)(row0 + arow) * Kd))[(kk >> 3) + ac4];
    }
    rb0 = ((const uint4*)(Wu + (size_t)(col0 + brow0) * Kd))[(kk >> 3) + bc4];
    rb1 = ((const uint4*)(Wu + (size_t)(col0 + brow1) * Kd))[(kk >> 3) + bc4];
  };

  v8f acc[4] = {};

  loadTiles(0);
  for (int kk = 0; kk < K; kk += 32) {
    ((uint4*)lA)[tid] = ra;
    ((uint4*)lB)[tid] = rb0;
    ((uint4*)lB)[tid + 256] = rb1;
    __syncthreads();

    if (kk + 32 < K) loadTiles(kk + 32);          // double-buffer into regs
    if (lane == 0 && kk + 64 < K)                  // prefetch 2 tiles ahead
      __builtin_prefetch((const void*)(Wu + (size_t)(col0 + wid * 16) * Kd + ((kk + 64) >> 1)), 0, 1);

    // fragment gather, contiguous k = 16*h + s for A and B (wave32)
    FragBF fa;
#pragma unroll
    for (int r = 0; r < 8; ++r) {
      int ka = r + 8 * h;                          // A: K {0-15}/{16-31}
      fa.u[r] = lA[(mi * 16 + lm) * 16 + ka];
    }
#pragma unroll
    for (int s = 0; s < 4; ++s) {
      FragBF fb;
#pragma unroll
      for (int r = 0; r < 8; ++r) {
        int kb = r + 8 * h;                        // B: K {0-15}/{16-31}
        fb.u[r] = lB[(nj * 64 + s * 16 + lm) * 16 + kb];
      }
      acc[s] = __builtin_amdgcn_wmma_f32_16x16x32_bf16(
          false, fa.v, false, fb.v, (short)0, acc[s], false, false);
    }
    __syncthreads();
  }

  // epilogue: C/D layout -> VGPR r holds (M = r + 8*h, N = lane&15)
#pragma unroll
  for (int s = 0; s < 4; ++s) {
#pragma unroll
    for (int r = 0; r < 8; ++r) {
      int gr = row0 + mi * 16 + r + 8 * h;
      int cc = col0 + nj * 64 + s * 16 + lm;
      float v = acc[s][r] + (bias ? bias[cc] : 0.0f);
      size_t idx = (size_t)gr * N + cc;
      if (mode == 0) {
        ((float*)Cout)[idx] = v;
      } else {
        v = v > 0.0f ? v : 0.0f;
        ((u16*)Cout)[idx] = f2bf(v * drop_scale(seed, (u32)idx, dropP));
      }
    }
  }
}

// ---------------- weight / input prep --------------------------------------
__global__ void kPackPad(u16* dst, const float* src, int N, int K, int Kp) {
  int i = blockIdx.x * blockDim.x + threadIdx.x;
  if (i >= N * Kp) return;
  int n = i / Kp, k = i % Kp;
  dst[i] = (k < K) ? f2bf(src[(size_t)n * K + k]) : (u16)0;
}
__global__ void kPackCat2(u16* dst, const float* a, int K1, const float* b, int K2, int N) {
  int Kp = K1 + K2;
  int i = blockIdx.x * blockDim.x + threadIdx.x;
  if (i >= N * Kp) return;
  int n = i / Kp, k = i % Kp;
  float v = (k < K1) ? a[(size_t)n * K1 + k] : b[(size_t)n * K2 + (k - K1)];
  dst[i] = f2bf(v);
}
__global__ void kPackProj(u16* dst, const float* pw, const float* gw) {
  int i = blockIdx.x * blockDim.x + threadIdx.x;
  if (i >= 128 * 1536) return;
  int n = i / 1536, k = i % 1536;
  float v = (n < M_) ? pw[(size_t)n * 1536 + k] : ((n == M_) ? gw[k] : 0.0f);
  dst[i] = f2bf(v);
}
__global__ void kAddBias(float* dst, const float* a, const float* b, int n) {
  int i = blockIdx.x * blockDim.x + threadIdx.x;
  if (i < n) dst[i] = a[i] + b[i];
}
__global__ void kPackBiasProj(float* dst, const float* pb, const float* gb) {
  int i = threadIdx.x;
  if (i < 128) dst[i] = (i < M_) ? pb[i] : ((i == M_) ? gb[0] : 0.0f);
}
__global__ void kBuildX(u16* dst, const float* din) {
  int i = blockIdx.x * blockDim.x + threadIdx.x;
  const int ROWS = (S_ + 1) * B_;
  if (i >= ROWS * 96) return;
  int row = i / 96, m = i % 96;
  int s = row / B_, b = row % B_;
  float v = (m < M_ && s > 0) ? din[((size_t)b * S_ + (s - 1)) * M_ + m] : 0.0f;
  dst[i] = f2bf(v);
}

// ---------------- recurrent-step kernels -----------------------------------
__global__ void kLstmCell(const float* __restrict__ gates, float* __restrict__ c,
                          u16* __restrict__ hbf, float p, u32 seed) {
  int i = blockIdx.x * blockDim.x + threadIdx.x;    // 64*1024 threads
  if (i >= B_ * H_) return;
  int b = i >> 10, j = i & (H_ - 1);
  const float* g = gates + (size_t)b * 4 * H_;
  float gi = g[j], gf = g[j + H_], gg = g[j + 2 * H_], go = g[j + 3 * H_];
  float cn = sigf(gf) * c[i] + sigf(gi) * tanhf(gg);
  float hn = sigf(go) * tanhf(cn);
  hn *= drop_scale(seed, (u32)i, p);
  c[i] = cn;
  hbf[i] = f2bf(hn);
}

__global__ void kConcatAttnIn(u16* dst, const u16* pnet_t, const float* ctx, const u16* hA) {
  int i = blockIdx.x * blockDim.x + threadIdx.x;
  if (i >= B_ * 1792) return;
  int b = i / 1792, j = i % 1792;
  u16 v;
  if (j < P_)            v = pnet_t[b * P_ + j];
  else if (j < P_ + E_)  v = f2bf(ctx[b * E_ + (j - P_)]);
  else                   v = hA[b * H_ + (j - P_ - E_)];
  dst[i] = v;
}
__global__ void kConcatDecIn(u16* dst, const u16* hA, const float* ctx, const u16* hD) {
  int i = blockIdx.x * blockDim.x + threadIdx.x;
  if (i >= B_ * 2560) return;
  int b = i / 2560, j = i % 2560;
  u16 v;
  if (j < H_)            v = hA[b * H_ + j];
  else if (j < H_ + E_)  v = f2bf(ctx[b * E_ + (j - H_)]);
  else                   v = hD[b * H_ + (j - H_ - E_)];
  dst[i] = v;
}
__global__ void kConcatHC(u16* dst, const u16* hD, const float* ctx) {
  int i = blockIdx.x * blockDim.x + threadIdx.x;
  if (i >= B_ * 1536) return;
  int b = i / 1536, j = i % 1536;
  dst[i] = (j < H_) ? hD[b * H_ + j] : f2bf(ctx[b * E_ + (j - H_)]);
}

// location conv + energies, LDS-staged; block = (256 t-values, one b)
__global__ __launch_bounds__(256) void kEnergies(
    const float* __restrict__ aw, const float* __restrict__ awc,
    const float* __restrict__ q, const float* __restrict__ pm,
    const float* __restrict__ convW, const float* __restrict__ ldw,
    const float* __restrict__ av, const int* __restrict__ lens,
    float* __restrict__ e) {
  __shared__ float sAw[T_], sAwc[T_];
  __shared__ float sLdw[A_ * FLOC];
  __shared__ float sCw[FLOC * 2 * KLOC];
  __shared__ float sQ[A_], sV[A_];

  const int b   = blockIdx.y;
  const int tid = threadIdx.x;
  for (int i = tid; i < T_; i += 256) { sAw[i] = aw[b * T_ + i]; sAwc[i] = awc[b * T_ + i]; }
  for (int i = tid; i < A_ * FLOC; i += 256) sLdw[i] = ldw[i];
  for (int i = tid; i < FLOC * 2 * KLOC; i += 256) sCw[i] = convW[i];
  if (tid < A_) { sQ[tid] = q[b * A_ + tid]; sV[tid] = av[tid]; }
  __syncthreads();

  const int t = blockIdx.x * 256 + tid;
  float locf[FLOC];
#pragma unroll 4
  for (int f = 0; f < FLOC; ++f) {
    float s = 0.0f;
    const float* w0 = sCw + f * 2 * KLOC;
    const float* w1 = w0 + KLOC;
    for (int k = 0; k < KLOC; ++k) {
      int tt = t + k - (KLOC - 1) / 2;
      if (tt >= 0 && tt < T_) s += w0[k] * sAw[tt] + w1[k] * sAwc[tt];
    }
    locf[f] = s;
  }
  float acc = 0.0f;
  const float* pmb = pm + ((size_t)b * T_ + t) * A_;
  for (int a = 0; a < A_; ++a) {
    float pl = 0.0f;
    const float* w = sLdw + a * FLOC;
#pragma unroll
    for (int f = 0; f < FLOC; ++f) pl += w[f] * locf[f];
    acc += sV[a] * tanhf(sQ[a] + pl + pmb[a]);
  }
  if (t >= lens[b]) acc = NEG_INF;
  e[b * T_ + t] = acc;
}

__global__ void kSoftmax(const float* __restrict__ e, float* __restrict__ aw,
                         float* __restrict__ awc, float* __restrict__ alignOut, int step) {
  __shared__ float sm[T_];
  int b = blockIdx.x, t = threadIdx.x;
  float x = e[b * T_ + t];
  sm[t] = x; __syncthreads();
  for (int s = T_ / 2; s > 0; s >>= 1) {
    if (t < s) sm[t] = fmaxf(sm[t], sm[t + s]);
    __syncthreads();
  }
  float mx = sm[0]; __syncthreads();
  float p = __expf(x - mx);
  sm[t] = p; __syncthreads();
  for (int s = T_ / 2; s > 0; s >>= 1) {
    if (t < s) sm[t] += sm[t + s];
    __syncthreads();
  }
  float pv = p / sm[0];
  aw[b * T_ + t] = pv;
  awc[b * T_ + t] += pv;
  alignOut[((size_t)b * S_ + step) * T_ + t] = pv;
}

// ctx[b,e] = sum_t aw[b,t] * memory[b,t,e]; block = (256 e-values, one b)
__global__ __launch_bounds__(256) void kCtx(const float* __restrict__ aw,
                                            const float* __restrict__ mem,
                                            float* __restrict__ ctx) {
  __shared__ float sAw[T_];
  const int b = blockIdx.y, tid = threadIdx.x;
  for (int i = tid; i < T_; i += 256) sAw[i] = aw[b * T_ + i];
  __syncthreads();
  const int eo = blockIdx.x * 256 + tid;
  float s = 0.0f;
  const float* m = mem + (size_t)b * T_ * E_ + eo;
  for (int t = 0; t < T_; ++t) s += sAw[t] * m[(size_t)t * E_];
  ctx[b * E_ + eo] = s;
}

__global__ void kScatterOut(const float* __restrict__ proj, float* __restrict__ out,
                            int step, size_t gateOff) {
  int i = blockIdx.x * blockDim.x + threadIdx.x;
  if (i >= B_ * (M_ + 1)) return;
  int b = i / (M_ + 1), m = i % (M_ + 1);
  float v = proj[b * 128 + m];
  if (m < M_) out[((size_t)b * S_ + step) * M_ + m] = v;
  else        out[gateOff + (size_t)b * S_ + step] = v;
}

// ---------------------------------------------------------------------------
extern "C" void kernel_launch(void* const* d_in, const int* in_sizes, int n_in,
                              void* d_out, int out_size, void* d_ws, size_t ws_size,
                              hipStream_t stream) {
  (void)in_sizes; (void)n_in; (void)out_size; (void)ws_size;

  const float* memory   = (const float*)d_in[0];
  const int*   memlen   = (const int*)d_in[1];
  const float* dec_in   = (const float*)d_in[2];
  const float* w1       = (const float*)d_in[3];
  const float* w2       = (const float*)d_in[4];
  const float* arW_ih   = (const float*)d_in[5];
  const float* arW_hh   = (const float*)d_in[6];
  const float* arB_ih   = (const float*)d_in[7];
  const float* arB_hh   = (const float*)d_in[8];
  const float* drW_ih   = (const float*)d_in[9];
  const float* drW_hh   = (const float*)d_in[10];
  const float* drB_ih   = (const float*)d_in[11];
  const float* drB_hh   = (const float*)d_in[12];
  const float* qW       = (const float*)d_in[13];
  const float* memW     = (const float*)d_in[14];
  const float* attnV    = (const float*)d_in[15];
  const float* convW    = (const float*)d_in[16];
  const float* ldW      = (const float*)d_in[17];
  const float* projW    = (const float*)d_in[18];
  const float* projB    = (const float*)d_in[19];
  const float* gateW    = (const float*)d_in[20];
  const float* gateB    = (const float*)d_in[21];
  float* out = (float*)d_out;

  // ---- workspace layout (256B aligned) ----
  char*  base = (char*)d_ws;
  size_t off  = 0;
  auto alloc = [&](size_t bytes) -> char* {
    char* p = base + off;
    off = (off + bytes + 255) & ~(size_t)255;
    return p;
  };
  const int ROWS = (S_ + 1) * B_;                       // 32064
  u16*   Wattn   = (u16*)  alloc((size_t)4096 * 1792 * 2);
  u16*   Wdec    = (u16*)  alloc((size_t)4096 * 2560 * 2);
  u16*   W1b     = (u16*)  alloc((size_t)P_ * 96 * 2);
  u16*   W2b     = (u16*)  alloc((size_t)P_ * P_ * 2);
  u16*   Wmem    = (u16*)  alloc((size_t)A_ * E_ * 2);
  u16*   Wq      = (u16*)  alloc((size_t)A_ * H_ * 2);
  u16*   Wproj   = (u16*)  alloc((size_t)128 * 1536 * 2);
  float* biasA   = (float*)alloc(4096 * 4);
  float* biasD   = (float*)alloc(4096 * 4);
  float* biasP   = (float*)alloc(128 * 4);
  float* pm      = (float*)alloc((size_t)B_ * T_ * A_ * 4);
  u16*   xpad    = (u16*)  alloc((size_t)ROWS * 96 * 2);
  u16*   abf1    = (u16*)  alloc((size_t)ROWS * P_ * 2);
  u16*   pnetbf  = (u16*)  alloc((size_t)ROWS * P_ * 2);
  // zero-initialized recurrent state (contiguous block)
  char*  state0  = base + off;
  float* ac      = (float*)alloc((size_t)B_ * H_ * 4);
  float* dc      = (float*)alloc((size_t)B_ * H_ * 4);
  u16*   hbfA    = (u16*)  alloc((size_t)B_ * H_ * 2);
  u16*   hbfD    = (u16*)  alloc((size_t)B_ * H_ * 2);
  float* aw      = (float*)alloc((size_t)B_ * T_ * 4);
  float* awc     = (float*)alloc((size_t)B_ * T_ * 4);
  float* ctx     = (float*)alloc((size_t)B_ * E_ * 4);
  size_t stateBytes = (size_t)((base + off) - state0);
  // per-step scratch
  u16*   xattn   = (u16*)  alloc((size_t)B_ * 1792 * 2);
  u16*   xdec    = (u16*)  alloc((size_t)B_ * 2560 * 2);
  u16*   hc      = (u16*)  alloc((size_t)B_ * 1536 * 2);
  float* gates   = (float*)alloc((size_t)B_ * 4096 * 4);
  float* qbuf    = (float*)alloc((size_t)B_ * A_ * 4);
  float* ener    = (float*)alloc((size_t)B_ * T_ * 4);
  float* projout = (float*)alloc((size_t)B_ * 128 * 4);

  const size_t gateOff  = (size_t)B_ * S_ * M_;
  const size_t alignOff = gateOff + (size_t)B_ * S_;

  // ---- one-time prep ----
  kAddBias<<<DIVUP(4096, 256), 256, 0, stream>>>(biasA, arB_ih, arB_hh, 4096);
  kAddBias<<<DIVUP(4096, 256), 256, 0, stream>>>(biasD, drB_ih, drB_hh, 4096);
  kPackBiasProj<<<1, 128, 0, stream>>>(biasP, projB, gateB);
  kPackPad<<<DIVUP(P_ * 96, 256), 256, 0, stream>>>(W1b, w1, P_, M_, 96);
  kPackPad<<<DIVUP(P_ * P_, 256), 256, 0, stream>>>(W2b, w2, P_, P_, P_);
  kPackPad<<<DIVUP(A_ * E_, 256), 256, 0, stream>>>(Wmem, memW, A_, E_, E_);
  kPackPad<<<DIVUP(A_ * H_, 256), 256, 0, stream>>>(Wq, qW, A_, H_, H_);
  kPackCat2<<<DIVUP(4096 * 1792, 256), 256, 0, stream>>>(Wattn, arW_ih, 768, arW_hh, H_, 4096);
  kPackCat2<<<DIVUP(4096 * 2560, 256), 256, 0, stream>>>(Wdec, drW_ih, 1536, drW_hh, H_, 4096);
  kPackProj<<<DIVUP(128 * 1536, 256), 256, 0, stream>>>(Wproj, projW, gateW);
  kBuildX<<<DIVUP(ROWS * 96, 256), 256, 0, stream>>>(xpad, dec_in);
  hipMemsetAsync(state0, 0, stateBytes, stream);

  // processed_memory = memory @ memW^T : (32768 x 512) x (128 x 512)^T
  kGemmT<1><<<dim3(B_ * T_ / 64, A_ / 128), 256, 0, stream>>>(
      memory, Wmem, nullptr, pm, E_, A_, 0, 0.0f, 0u);
  // prenet layer1: relu+dropout(0.5) -> bf16
  kGemmT<0><<<dim3(ROWS / 64, P_ / 128), 256, 0, stream>>>(
      xpad, W1b, nullptr, abf1, 96, P_, 1, 0.5f, 101u);
  // prenet layer2: relu+dropout(0.5) -> bf16
  kGemmT<0><<<dim3(ROWS / 64, P_ / 128), 256, 0, stream>>>(
      abf1, W2b, nullptr, pnetbf, P_, P_, 1, 0.5f, 102u);

  // ---- 500 sequential decoder steps ----
  for (int t = 0; t < S_; ++t) {
    const u16* pnet_t = pnetbf + (size_t)t * B_ * P_;

    kConcatAttnIn<<<DIVUP(B_ * 1792, 256), 256, 0, stream>>>(xattn, pnet_t, ctx, hbfA);
    kGemmT<0><<<dim3(1, 4096 / 128), 256, 0, stream>>>(
        xattn, Wattn, biasA, gates, 1792, 4096, 0, 0.0f, 0u);
    kLstmCell<<<DIVUP(B_ * H_, 256), 256, 0, stream>>>(gates, ac, hbfA, 0.1f, 1000u + 2u * t);

    kGemmT<0><<<dim3(1, A_ / 128), 256, 0, stream>>>(
        hbfA, Wq, nullptr, qbuf, H_, A_, 0, 0.0f, 0u);
    kEnergies<<<dim3(T_ / 256, B_), 256, 0, stream>>>(
        aw, awc, qbuf, pm, convW, ldW, attnV, memlen, ener);
    kSoftmax<<<B_, T_, 0, stream>>>(ener, aw, awc, out + alignOff, t);
    kCtx<<<dim3(E_ / 256, B_), 256, 0, stream>>>(aw, memory, ctx);

    kConcatDecIn<<<DIVUP(B_ * 2560, 256), 256, 0, stream>>>(xdec, hbfA, ctx, hbfD);
    kGemmT<0><<<dim3(1, 4096 / 128), 256, 0, stream>>>(
        xdec, Wdec, biasD, gates, 2560, 4096, 0, 0.0f, 0u);
    kLstmCell<<<DIVUP(B_ * H_, 256), 256, 0, stream>>>(gates, dc, hbfD, 0.1f, 1000u + 2u * t + 1u);

    kConcatHC<<<DIVUP(B_ * 1536, 256), 256, 0, stream>>>(hc, hbfD, ctx);
    kGemmT<0><<<dim3(1, 128 / 128), 256, 0, stream>>>(
        hc, Wproj, biasP, projout, 1536, 128, 0, 0.0f, 0u);
    kScatterOut<<<DIVUP(B_ * (M_ + 1), 128), 128, 0, stream>>>(projout, out, t, gateOff);
  }
}
